// CrossAttention3D_48945447305765
// MI455X (gfx1250) — matrix operand order repaired and checked by reading, not compiled, not run on hardware
//
#include <hip/hip_runtime.h>

typedef _Float16 h16;
typedef __attribute__((ext_vector_type(16))) _Float16 v16h;
typedef __attribute__((ext_vector_type(8)))  _Float16 v8h;
typedef __attribute__((ext_vector_type(8)))  float    v8f;

#define BATCH 8
#define CDIM  256
#define CTXT  768
#define NTOK  13824
#define NTILES 864
#define LTXT  77
#define LK    80      // kh rows padded to 5*16
#define LV    96      // vhT cols / attn cols padded to 3*32
#define HEADS 8
#define HD    32
#define INV_SQRT_HD 0.17677669529663687f

static __device__ __forceinline__ v8f wmma_f16(v16h a, v16h b, v8f c) {
  return __builtin_amdgcn_wmma_f32_16x16x32_f16(false, a, false, b, (short)0, c, false, false);
}

// A-fragment (16x32 f16, M=row per lane) gathered from a row-major f16 LDS row.
// Lane layout: halves 0-15 -> K = k0 + grp*16 + j (contiguous, matches B-fragment loads)
static __device__ __forceinline__ v16h load_afrag(const h16* row, int k0, int grp) {
  union { v16h v; v8h h[2]; } u;
  u.h[0] = *(const v8h*)(row + k0 + grp * 16);
  u.h[1] = *(const v8h*)(row + k0 + grp * 16 + 8);
  return u.v;
}

// ---------------- weight fusion kernels (tiny) ----------------

__global__ __launch_bounds__(256) void fuse_qo_kernel(
    const float* __restrict__ q_proj_w, const float* __restrict__ in_q_w,
    const float* __restrict__ out_proj_w, const float* __restrict__ out_conv_w,
    h16* __restrict__ Wq, h16* __restrict__ Wo) {
  int idx = blockIdx.x * 256 + threadIdx.x;      // 0 .. 131071
  int o = (idx >> 8) & 255;
  int c = idx & 255;
  if (idx < 65536) {
    float s = 0.f;
    for (int j = 0; j < 256; ++j) s += in_q_w[o * 256 + j] * q_proj_w[j * 256 + c];
    Wq[idx] = (h16)(s * INV_SQRT_HD);
  } else {
    float s = 0.f;
    for (int j = 0; j < 256; ++j) s += out_conv_w[o * 256 + j] * out_proj_w[j * 256 + c];
    Wo[idx - 65536] = (h16)s;
  }
}

__global__ void fuse_bias_kernel(const float* __restrict__ in_q_b,
                                 const float* __restrict__ out_conv_w,
                                 const float* __restrict__ out_proj_b,
                                 const float* __restrict__ out_conv_b,
                                 float* __restrict__ bq, float* __restrict__ bo) {
  int i = blockIdx.x * 256 + threadIdx.x;
  if (i < 256) {
    bq[i] = in_q_b[i] * INV_SQRT_HD;
  } else if (i < 512) {
    int o = i - 256;
    float s = out_conv_b[o];
    for (int j = 0; j < 256; ++j) s += out_conv_w[o * 256 + j] * out_proj_b[j];
    bo[o] = s;
  }
}

__global__ __launch_bounds__(256) void fuse_kv_kernel(
    const float* __restrict__ k_proj_w, const float* __restrict__ v_proj_w,
    const float* __restrict__ in_k_w, const float* __restrict__ in_v_w,
    float* __restrict__ Wk, float* __restrict__ Wv) {
  int idx = blockIdx.x * 256 + threadIdx.x;      // 0 .. 393215
  int which = (idx >= 196608) ? 1 : 0;
  int e = which ? idx - 196608 : idx;
  int c = e / CTXT, t = e % CTXT;
  const float* inw = which ? in_v_w : in_k_w;
  const float* pw  = which ? v_proj_w : k_proj_w;
  float s = 0.f;
  for (int j = 0; j < 256; ++j) s += inw[c * 256 + j] * pw[j * CTXT + t];
  (which ? Wv : Wk)[e] = s;
}

// kh: [B][H][LK][HD] f16 (zero padded rows); vhT: [B][H][HD][LV] f16 (zero padded cols)
__global__ __launch_bounds__(256) void kv_kernel(
    const float* __restrict__ x_txt, const float* __restrict__ Wk, const float* __restrict__ Wv,
    const float* __restrict__ in_k_b, const float* __restrict__ in_v_b,
    h16* __restrict__ kh, h16* __restrict__ vhT) {
  int idx = blockIdx.x * 256 + threadIdx.x;      // 0 .. 360447
  const int NKH = BATCH * HEADS * LK * HD;       // 163840
  if (idx < NKH) {
    int d = idx & 31;
    int l = (idx >> 5) % LK;
    int h = (idx / (32 * LK)) & 7;
    int b = idx / (32 * LK * HEADS);
    int c = h * HD + d;
    float s = 0.f;
    if (l < LTXT) {
      const float* xr = x_txt + ((size_t)b * LTXT + l) * CTXT;
      const float* wr = Wk + (size_t)c * CTXT;
      s = in_k_b[c];
      for (int t = 0; t < CTXT; ++t) s += xr[t] * wr[t];
    }
    kh[idx] = (h16)s;
  } else {
    int j = idx - NKH;
    int l = j % LV;
    int d = (j / LV) & 31;
    int h = (j / (LV * 32)) & 7;
    int b = j / (LV * 32 * HEADS);
    int c = h * HD + d;
    float s = 0.f;
    if (l < LTXT) {
      const float* xr = x_txt + ((size_t)b * LTXT + l) * CTXT;
      const float* wr = Wv + (size_t)c * CTXT;
      s = in_v_b[c];
      for (int t = 0; t < CTXT; ++t) s += xr[t] * wr[t];
    }
    vhT[j] = (h16)s;
  }
}

__global__ void zero_kernel(float* __restrict__ p, int n) {
  int i = blockIdx.x * 256 + threadIdx.x;
  if (i < n) p[i] = 0.f;
}

// ------- fused attention kernel: one wave per PAIR of 16-token tiles -------
// Two tiles share every Wq/Wo B-fragment load (halves L2 weight traffic,
// doubles WMMA issued per vector load).

__global__ __launch_bounds__(32) void attn_kernel(
    const float* __restrict__ x_img, const unsigned char* __restrict__ txt_mask,
    const h16* __restrict__ Wq, const h16* __restrict__ Wo,
    const float* __restrict__ bq, const float* __restrict__ bo,
    const h16* __restrict__ kh, const h16* __restrict__ vhT,
    float* __restrict__ rbuf, float* __restrict__ gsum, float* __restrict__ gssq) {
  const int pair = blockIdx.x;            // 0 .. B*NTILES/2-1
  const int b = pair / (NTILES / 2);
  const int n0 = (pair % (NTILES / 2)) * 32;
  const int lane = threadIdx.x;
  const int grp = lane >> 4;              // 0 or 1
  const int col = lane & 15;

  __shared__ alignas(64) h16 sh_o[2][16 * 256];  // per-tile attention output tiles
  __shared__ alignas(64) h16 sh_q[2][16 * 32];   // per-tile per-head qh tiles
  __shared__ alignas(64) h16 sh_p[16 * 96];      // attn probs (cols 80..95 stay zero)

  for (int i = lane; i < 16 * 96; i += 32) sh_p[i] = (h16)0.f;

  const float* xb = x_img + (size_t)b * CDIM * NTOK;

  // Build x A-fragments for both token tiles directly from global memory.
  // For a fixed half-index, lanes 0-15 read 16 consecutive tokens of one
  // channel (one 64B segment), lanes 16-31 another channel: fully coalesced.
  v16h ax[2][8];
  #pragma unroll
  for (int tt = 0; tt < 2; ++tt) {
    const int n = n0 + tt * 16 + col;
    #pragma unroll
    for (int kc = 0; kc < 8; ++kc) {
      union { v16h v; h16 e[16]; } u;
      #pragma unroll
      for (int j = 0; j < 8; ++j) {
        int c0 = kc * 32 + grp * 16 + j;
        u.e[j]     = (h16)xb[(size_t)c0 * NTOK + n];
        u.e[8 + j] = (h16)xb[(size_t)(c0 + 8) * NTOK + n];
      }
      ax[tt][kc] = u.v;
    }
  }

  bool mb[5];
  #pragma unroll
  for (int lb = 0; lb < 5; ++lb) {
    int cg = lb * 16 + col;
    mb[lb] = (cg >= LTXT) || (txt_mask[b * LTXT + (cg < LTXT ? cg : 0)] != 0);
  }

  const h16* khb = kh  + (size_t)b * HEADS * LK * HD;
  const h16* vtb = vhT + (size_t)b * HEADS * HD * LV;

  for (int h = 0; h < HEADS; ++h) {
    // ---- qh = x * Wq^T + bq for both tiles (B-fragments loaded once) ----
    #pragma unroll
    for (int nb = 0; nb < 2; ++nb) {
      int ch = h * 32 + nb * 16 + col;
      float bias = bq[ch];
      v8f a0, a1;
      #pragma unroll
      for (int i = 0; i < 8; ++i) { a0[i] = bias; a1[i] = bias; }
      const h16* wrow = Wq + (size_t)ch * 256;   // row-major: lane = column N, K contiguous
      #pragma unroll
      for (int kc = 0; kc < 8; ++kc) {
        v16h bf = *(const v16h*)(wrow + kc * 32 + grp * 16);
        a0 = wmma_f16(ax[0][kc], bf, a0);
        a1 = wmma_f16(ax[1][kc], bf, a1);
      }
      #pragma unroll
      for (int rr = 0; rr < 8; ++rr) {
        sh_q[0][(rr + 8 * grp) * 32 + nb * 16 + col] = (h16)a0[rr];
        sh_q[1][(rr + 8 * grp) * 32 + nb * 16 + col] = (h16)a1[rr];
      }
    }
    __syncthreads();

    const h16* khh = khb + (size_t)h * LK * HD;
    const h16* vth = vtb + (size_t)h * HD * LV;

    for (int tt = 0; tt < 2; ++tt) {
      v16h aq = load_afrag(&sh_q[tt][col * 32], 0, grp);

      // ---- scores = qh * kh^T (scale folded into Wq) ----
      v8f sc[5];
      #pragma unroll
      for (int lb = 0; lb < 5; ++lb) {
        v8f z;
        #pragma unroll
        for (int i = 0; i < 8; ++i) z[i] = 0.f;
        sc[lb] = wmma_f16(aq, *(const v16h*)(khh + (size_t)(lb * 16 + col) * HD + grp * 16), z);
      }

      // ---- mask + softmax over 77 keys ----
      #pragma unroll
      for (int lb = 0; lb < 5; ++lb)
        if (mb[lb]) {
          #pragma unroll
          for (int rr = 0; rr < 8; ++rr) sc[lb][rr] = -1e30f;
        }

      float mx[8], sm[8];
      #pragma unroll
      for (int rr = 0; rr < 8; ++rr) {
        float m = sc[0][rr];
        #pragma unroll
        for (int lb = 1; lb < 5; ++lb) m = fmaxf(m, sc[lb][rr]);
        mx[rr] = m;
      }
      #pragma unroll
      for (int off = 1; off <= 8; off <<= 1) {
        #pragma unroll
        for (int rr = 0; rr < 8; ++rr) mx[rr] = fmaxf(mx[rr], __shfl_xor(mx[rr], off, 32));
      }
      #pragma unroll
      for (int rr = 0; rr < 8; ++rr) {
        float s = 0.f;
        #pragma unroll
        for (int lb = 0; lb < 5; ++lb) {
          float e = __expf(sc[lb][rr] - mx[rr]);
          sc[lb][rr] = e;
          s += e;
        }
        sm[rr] = s;
      }
      #pragma unroll
      for (int off = 1; off <= 8; off <<= 1) {
        #pragma unroll
        for (int rr = 0; rr < 8; ++rr) sm[rr] += __shfl_xor(sm[rr], off, 32);
      }
      #pragma unroll
      for (int rr = 0; rr < 8; ++rr) {
        float inv = 1.0f / sm[rr];
        #pragma unroll
        for (int lb = 0; lb < 5; ++lb)
          sh_p[(rr + 8 * grp) * 96 + lb * 16 + col] = (h16)(sc[lb][rr] * inv);
      }
      __syncthreads();

      // ---- o_head = attn * vh (vhT zero-padded beyond 77) ----
      v16h ap[3];
      #pragma unroll
      for (int kc = 0; kc < 3; ++kc) ap[kc] = load_afrag(&sh_p[col * 96], kc * 32, grp);
      #pragma unroll
      for (int nb = 0; nb < 2; ++nb) {
        v8f acc;
        #pragma unroll
        for (int i = 0; i < 8; ++i) acc[i] = 0.f;
        const h16* vrow = vth + (size_t)(nb * 16 + col) * LV;
        #pragma unroll
        for (int kc = 0; kc < 3; ++kc)
          acc = wmma_f16(ap[kc], *(const v16h*)(vrow + kc * 32 + grp * 16), acc);
        #pragma unroll
        for (int rr = 0; rr < 8; ++rr)
          sh_o[tt][(rr + 8 * grp) * 256 + h * 32 + nb * 16 + col] = (h16)acc[rr];
      }
      __syncthreads();
    }
  }

  // ---- fused out_proj+out_conv, residual, instance-norm partial sums ----
  v16h ao[2][8];
  #pragma unroll
  for (int tt = 0; tt < 2; ++tt) {
    #pragma unroll
    for (int kc = 0; kc < 8; ++kc)
      ao[tt][kc] = load_afrag(&sh_o[tt][col * 256], kc * 32, grp);
  }

  for (int nb = 0; nb < 16; ++nb) {
    int ch = nb * 16 + col;
    float bias = bo[ch];
    v8f a0, a1;
    #pragma unroll
    for (int i = 0; i < 8; ++i) { a0[i] = bias; a1[i] = bias; }
    const h16* wrow = Wo + (size_t)ch * 256;
    #pragma unroll
    for (int kc = 0; kc < 8; ++kc) {
      v16h bf = *(const v16h*)(wrow + kc * 32 + grp * 16);
      a0 = wmma_f16(ao[0][kc], bf, a0);
      a1 = wmma_f16(ao[1][kc], bf, a1);
    }

    const float* xcol = xb + (size_t)ch * NTOK + n0;
    float* rcol = rbuf + ((size_t)b * CDIM + ch) * NTOK + n0;
    #pragma unroll
    for (int tt = 0; tt < 2; ++tt) {
      const v8f& acc = tt ? a1 : a0;
      float s1 = 0.f, s2 = 0.f;
      #pragma unroll
      for (int rr = 0; rr < 8; ++rr) {
        float rv = acc[rr] + xcol[tt * 16 + rr + 8 * grp];
        rcol[tt * 16 + rr + 8 * grp] = rv;
        s1 += rv;
        s2 += rv * rv;
      }
      s1 += __shfl_xor(s1, 16, 32);
      s2 += __shfl_xor(s2, 16, 32);
      if (grp == 0) {
        atomicAdd(&gsum[b * CDIM + ch], s1);
        atomicAdd(&gssq[b * CDIM + ch], s2);
      }
    }
  }
}

// ---------------- epilogue ----------------

__global__ void stats_final_kernel(const float* __restrict__ gsum, const float* __restrict__ gssq,
                                   const float* __restrict__ norm_w, const float* __restrict__ norm_b,
                                   float* __restrict__ scl, float* __restrict__ sft) {
  int i = blockIdx.x * 256 + threadIdx.x;
  if (i >= BATCH * CDIM) return;
  int c = i & 255;
  float inv_n = 1.0f / (float)NTOK;
  float mu = gsum[i] * inv_n;
  float var = gssq[i] * inv_n - mu * mu;
  float rs = rsqrtf(var + 1e-5f);
  float a = norm_w[c] * rs;
  scl[i] = a;
  sft[i] = norm_b[c] - mu * a;
}

__global__ __launch_bounds__(256) void norm_silu_kernel(
    const float* __restrict__ rbuf, const float* __restrict__ scl,
    const float* __restrict__ sft, float* __restrict__ out, unsigned int total) {
  unsigned int idx = blockIdx.x * 256 + threadIdx.x;
  if (idx >= total) return;
  int bc = (int)(idx / NTOK);
  float v = rbuf[idx] * scl[bc] + sft[bc];
  out[idx] = v / (1.0f + __expf(-v));
}

// ---------------- launcher ----------------

extern "C" void kernel_launch(void* const* d_in, const int* in_sizes, int n_in,
                              void* d_out, int out_size, void* d_ws, size_t ws_size,
                              hipStream_t stream) {
  const float* x_img       = (const float*)d_in[0];
  const float* x_txt       = (const float*)d_in[1];
  const unsigned char* txt_mask = (const unsigned char*)d_in[2];
  const float* q_proj_w    = (const float*)d_in[3];
  const float* k_proj_w    = (const float*)d_in[4];
  const float* v_proj_w    = (const float*)d_in[5];
  const float* in_q_w      = (const float*)d_in[6];
  const float* in_q_b      = (const float*)d_in[7];
  const float* in_k_w      = (const float*)d_in[8];
  const float* in_k_b      = (const float*)d_in[9];
  const float* in_v_w      = (const float*)d_in[10];
  const float* in_v_b      = (const float*)d_in[11];
  const float* out_proj_w  = (const float*)d_in[12];
  const float* out_proj_b  = (const float*)d_in[13];
  const float* out_conv_w  = (const float*)d_in[14];
  const float* out_conv_b  = (const float*)d_in[15];
  const float* norm_w      = (const float*)d_in[16];
  const float* norm_b      = (const float*)d_in[17];

  char* ws = (char*)d_ws;
  size_t off = 0;
  auto alloc = [&](size_t bytes) -> void* {
    void* p = ws + off;
    off += (bytes + 255) & ~(size_t)255;
    return p;
  };
  h16*   Wq    = (h16*)  alloc((size_t)65536 * 2);
  h16*   Wo    = (h16*)  alloc((size_t)65536 * 2);
  float* bq    = (float*)alloc(256 * 4);
  float* bo    = (float*)alloc(256 * 4);
  float* Wk    = (float*)alloc((size_t)CDIM * CTXT * 4);
  float* Wv    = (float*)alloc((size_t)CDIM * CTXT * 4);
  h16*   kh    = (h16*)  alloc((size_t)BATCH * HEADS * LK * HD * 2);
  h16*   vhT   = (h16*)  alloc((size_t)BATCH * HEADS * HD * LV * 2);
  float* gsum  = (float*)alloc(BATCH * CDIM * 4);
  float* gssq  = (float*)alloc(BATCH * CDIM * 4);
  float* scl   = (float*)alloc(BATCH * CDIM * 4);
  float* sft   = (float*)alloc(BATCH * CDIM * 4);
  float* rbuf  = (float*)alloc((size_t)BATCH * CDIM * NTOK * 4);

  float* out = (float*)d_out;

  fuse_qo_kernel<<<512, 256, 0, stream>>>(q_proj_w, in_q_w, out_proj_w, out_conv_w, Wq, Wo);
  fuse_bias_kernel<<<2, 256, 0, stream>>>(in_q_b, out_conv_w, out_proj_b, out_conv_b, bq, bo);
  fuse_kv_kernel<<<1536, 256, 0, stream>>>(k_proj_w, v_proj_w, in_k_w, in_v_w, Wk, Wv);
  kv_kernel<<<1408, 256, 0, stream>>>(x_txt, Wk, Wv, in_k_b, in_v_b, kh, vhT);
  zero_kernel<<<16, 256, 0, stream>>>(gsum, BATCH * CDIM);
  zero_kernel<<<16, 256, 0, stream>>>(gssq, BATCH * CDIM);
  attn_kernel<<<BATCH * NTILES / 2, 32, 0, stream>>>(x_img, txt_mask, Wq, Wo, bq, bo,
                                                     kh, vhT, rbuf, gsum, gssq);
  stats_final_kernel<<<8, 256, 0, stream>>>(gsum, gssq, norm_w, norm_b, scl, sft);
  unsigned int total = (unsigned int)BATCH * CDIM * NTOK;
  norm_silu_kernel<<<(total + 255) / 256, 256, 0, stream>>>(rbuf, scl, sft, out, total);
}
